// AGCRN_47064251629613
// MI455X (gfx1250) — matrix operand order repaired and checked by reading, not compiled, not run on hardware
//
#include <hip/hip_runtime.h>
#include <hip/hip_bf16.h>

// Problem constants (from reference)
#define BB 32
#define TT 12
#define NN 2048
#define DD 10
#define HH 32
#define CIN 33          // C + H = 1 + 32
#define NCOL (BB * CIN) // 1056 GEMM columns
#define GW_E (2 * CIN * 64) // 4224 per-node gate weight elems
#define UW_E (2 * CIN * 32) // 2112 per-node update weight elems

typedef __attribute__((ext_vector_type(16))) _Float16 v16h;
typedef __attribute__((ext_vector_type(8)))  _Float16 v8h;
typedef __attribute__((ext_vector_type(8)))  float    v8f;

// ---------------------------------------------------------------------------
// A = softmax(relu(E @ E^T)) row-wise, stored f16 for WMMA consumption.
// One block per row n.
// ---------------------------------------------------------------------------
__global__ __launch_bounds__(256) void adj_kernel(const float* __restrict__ E,
                                                  _Float16* __restrict__ A16) {
  __shared__ float En[16];
  __shared__ float red[256];
  const int n = blockIdx.x;
  const int tid = threadIdx.x;
  if (tid < DD) En[tid] = E[n * DD + tid];
  __syncthreads();

  // pass 1: row max of relu(dot)
  float lmax = 0.0f; // relu() >= 0
  for (int m = tid; m < NN; m += 256) {
    float d = 0.f;
#pragma unroll
    for (int j = 0; j < DD; ++j) d += En[j] * E[m * DD + j];
    lmax = fmaxf(lmax, fmaxf(d, 0.f));
  }
  red[tid] = lmax;
  __syncthreads();
  for (int s = 128; s > 0; s >>= 1) {
    if (tid < s) red[tid] = fmaxf(red[tid], red[tid + s]);
    __syncthreads();
  }
  const float rowmax = red[0];
  __syncthreads();

  // pass 2: sum of exp
  float lsum = 0.f;
  for (int m = tid; m < NN; m += 256) {
    float d = 0.f;
#pragma unroll
    for (int j = 0; j < DD; ++j) d += En[j] * E[m * DD + j];
    d = fmaxf(d, 0.f);
    lsum += __expf(d - rowmax);
  }
  red[tid] = lsum;
  __syncthreads();
  for (int s = 128; s > 0; s >>= 1) {
    if (tid < s) red[tid] += red[tid + s];
    __syncthreads();
  }
  const float inv = 1.0f / red[0];
  __syncthreads();

  // pass 3: write normalized f16
  for (int m = tid; m < NN; m += 256) {
    float d = 0.f;
#pragma unroll
    for (int j = 0; j < DD; ++j) d += En[j] * E[m * DD + j];
    d = fmaxf(d, 0.f);
    A16[(size_t)n * NN + m] = (_Float16)(__expf(d - rowmax) * inv);
  }
}

// ---------------------------------------------------------------------------
// Per-node weights/biases from embedding pools. One block per node.
// ---------------------------------------------------------------------------
__global__ __launch_bounds__(256) void params_kernel(
    const float* __restrict__ E, const float* __restrict__ gW,
    const float* __restrict__ gb, const float* __restrict__ uW,
    const float* __restrict__ ub, float* __restrict__ Wg,
    float* __restrict__ bg, float* __restrict__ Wu, float* __restrict__ bu) {
  __shared__ float En[16];
  const int n = blockIdx.x;
  const int tid = threadIdx.x;
  if (tid < DD) En[tid] = E[n * DD + tid];
  __syncthreads();
  for (int e = tid; e < GW_E; e += 256) {
    float a = 0.f;
#pragma unroll
    for (int d = 0; d < DD; ++d) a += En[d] * gW[d * GW_E + e];
    Wg[(size_t)n * GW_E + e] = a;
  }
  for (int e = tid; e < UW_E; e += 256) {
    float a = 0.f;
#pragma unroll
    for (int d = 0; d < DD; ++d) a += En[d] * uW[d * UW_E + e];
    Wu[(size_t)n * UW_E + e] = a;
  }
  if (tid < 64) {
    float a = 0.f;
#pragma unroll
    for (int d = 0; d < DD; ++d) a += En[d] * gb[d * 64 + tid];
    bg[n * 64 + tid] = a;
  }
  if (tid < 32) {
    float a = 0.f;
#pragma unroll
    for (int d = 0; d < DD; ++d) a += En[d] * ub[d * 32 + tid];
    bu[n * 32 + tid] = a;
  }
}

__global__ void zero_kernel(float* __restrict__ p, int n) {
  int i = blockIdx.x * blockDim.x + threadIdx.x;
  if (i < n) p[i] = 0.f;
}

// ---------------------------------------------------------------------------
// Pack X TRANSPOSED: X16T[col][m], col = b*33+c.  Variant 1: concat([x_t, h]);
// variant 2: concat([x_t, z*h]).  Transposed layout makes the GEMM B-fragment
// gather contiguous (two ds_load_b128 per lane instead of 16 ds_load_u16).
// ---------------------------------------------------------------------------
__global__ void build_xs1(const float* __restrict__ src,
                          const float* __restrict__ hin,
                          _Float16* __restrict__ XT, int t) {
  int idx = blockIdx.x * blockDim.x + threadIdx.x;
  if (idx >= NN * NCOL) return;
  int m = idx % NN, col = idx / NN;
  int b = col / CIN, c = col % CIN;
  float v = (c == 0) ? src[(b * TT + t) * NN + m]
                     : hin[((size_t)b * NN + m) * HH + c - 1];
  XT[idx] = (_Float16)v;  // idx == col*NN + m
}

__global__ void build_xs2(const float* __restrict__ src,
                          const float* __restrict__ hin,
                          const float* __restrict__ zr,
                          _Float16* __restrict__ XT, int t) {
  int idx = blockIdx.x * blockDim.x + threadIdx.x;
  if (idx >= NN * NCOL) return;
  int m = idx % NN, col = idx / NN;
  int b = col / CIN, c = col % CIN;
  float v;
  if (c == 0) v = src[(b * TT + t) * NN + m];
  else {
    size_t base = (size_t)b * NN + m;
    v = zr[base * 64 + c - 1] * hin[base * HH + c - 1];  // z * h
  }
  XT[idx] = (_Float16)v;
}

// ---------------------------------------------------------------------------
// WMMA GEMM: Y[2048,1056](f32) = A16[2048,2048] @ X16T^T
//   A16 : row-major [M=2048][K=2048] f16
//   XT  : [Ncol=1056][K=2048] f16 (B operand pre-transposed -> contiguous K)
// Block = 128 threads = 4 waves; block tile 64(M) x 32(N); each wave computes
// a 16x32 strip = 2 accumulators, reusing one A fragment -> 2 WMMA / K-step.
// Fragment packing per CDNA5 ISA 7.12.2 16-bit layouts.
// ---------------------------------------------------------------------------
__global__ __launch_bounds__(128) void gemm_kernel(
    const _Float16* __restrict__ A, const _Float16* __restrict__ XT,
    float* __restrict__ Y) {
  __shared__ _Float16 As[64][40];  // +8 pad keeps 16B alignment per row
  __shared__ _Float16 Xs[32][40];  // Xs[col][k]
  const int tid = threadIdx.x;
  const int lane = tid & 31;
  const int lm = lane & 15;
  const int lh = lane >> 4;        // 0: lanes 0-15, 1: lanes 16-31
  const int wave = tid >> 5;       // 0..3 -> M subtile
  const int rowBase = blockIdx.x * 64;
  const int colBase = blockIdx.y * 32;
  const int lr = tid >> 2;         // 0..31 staging row
  const int lc = (tid & 3) * 8;    // 0,8,16,24 staging col (8 halves = b128)

  union V16 { v16h v; v8h h[2]; };

  v8f acc0 = {}, acc1 = {};
  for (int k0 = 0; k0 < NN; k0 += 32) {
    // stage 64x32 A tile and 32(cols)x32(k) X tile (both K-contiguous rows)
    *(v8h*)&As[lr][lc] =
        *(const v8h*)&A[(size_t)(rowBase + lr) * NN + k0 + lc];
    *(v8h*)&As[lr + 32][lc] =
        *(const v8h*)&A[(size_t)(rowBase + lr + 32) * NN + k0 + lc];
    *(v8h*)&Xs[lr][lc] =
        *(const v8h*)&XT[(size_t)(colBase + lr) * NN + k0 + lc];
    if (k0 + 32 < NN) {  // hint next K tile (global_prefetch_b8)
      __builtin_prefetch(&A[(size_t)(rowBase + lr) * NN + k0 + 32 + lc], 0, 1);
      __builtin_prefetch(&XT[(size_t)(colBase + lr) * NN + k0 + 32 + lc], 0, 1);
    }
    __syncthreads();

    // A fragment: lane holds row M=lm; lanes<16: K 0-7 & 16-23; lanes>=16: +8
    V16 a, b0, b1;
    const int arow = wave * 16 + lm;
    const int kb = lh * 8;
    a.h[0] = *(const v8h*)&As[arow][kb];
    a.h[1] = *(const v8h*)&As[arow][16 + kb];
    // B fragment: lane holds col N=lm; lanes<16: K 0-7 & 16-23; lanes>=16: +8
    const int kbb = lh * 8;
    b0.h[0] = *(const v8h*)&Xs[lm][kbb];
    b0.h[1] = *(const v8h*)&Xs[lm][kbb + 16];
    b1.h[0] = *(const v8h*)&Xs[16 + lm][kbb];
    b1.h[1] = *(const v8h*)&Xs[16 + lm][kbb + 16];

    acc0 = __builtin_amdgcn_wmma_f32_16x16x32_f16(
        false, a.v, false, b0.v, (short)0, acc0, false, false);
    acc1 = __builtin_amdgcn_wmma_f32_16x16x32_f16(
        false, a.v, false, b1.v, (short)0, acc1, false, false);
    __syncthreads();
  }

  // C/D layout: VGPR r, lanes<16 -> M=r, lanes>=16 -> M=r+8; N=lane%16
  const int outRow = rowBase + wave * 16 + lh * 8;
#pragma unroll
  for (int r = 0; r < 8; ++r) {
    Y[(size_t)(outRow + r) * NCOL + colBase + lm]      = acc0[r];
    Y[(size_t)(outRow + r) * NCOL + colBase + 16 + lm] = acc1[r];
  }
}

// ---------------------------------------------------------------------------
// zr = sigmoid( sum_i xs_i*Wg[n,0,i,o] + Y_i*Wg[n,1,i,o] + bg )  (o in 0..63)
// ---------------------------------------------------------------------------
__global__ __launch_bounds__(256) void gate_kernel(
    const float* __restrict__ src, const float* __restrict__ hin,
    const float* __restrict__ Y, const float* __restrict__ Wg,
    const float* __restrict__ bg, float* __restrict__ zr, int t) {
  int idx = blockIdx.x * blockDim.x + threadIdx.x;
  if (idx >= BB * NN * 64) return;
  int o = idx & 63;
  int n = (idx >> 6) & (NN - 1);
  int b = idx >> 17;
  const float* w = Wg + (size_t)n * GW_E;
  const float* yrow = Y + (size_t)n * NCOL + b * CIN;
  float acc = bg[n * 64 + o];
  float x0 = src[(b * TT + t) * NN + n];
  acc += x0 * w[o] + yrow[0] * w[2112 + o];  // 2112 == k-stride of Wg
  size_t hb = ((size_t)b * NN + n) * HH;
#pragma unroll 4
  for (int i = 1; i < CIN; ++i) {
    float xi = hin[hb + i - 1];
    acc += xi * w[i * 64 + o] + yrow[i] * w[2112 + i * 64 + o];
  }
  zr[idx] = 1.f / (1.f + __expf(-acc));
}

// ---------------------------------------------------------------------------
// hc = tanh(gconv2); h_out = r*h + (1-r)*hc   (o in 0..31)
// ---------------------------------------------------------------------------
__global__ __launch_bounds__(256) void update_kernel(
    const float* __restrict__ src, const float* __restrict__ hin,
    const float* __restrict__ zr, const float* __restrict__ Y,
    const float* __restrict__ Wu, const float* __restrict__ bu,
    float* __restrict__ hout, int t) {
  int idx = blockIdx.x * blockDim.x + threadIdx.x;
  if (idx >= BB * NN * HH) return;
  int o = idx & 31;
  int n = (idx >> 5) & (NN - 1);
  int b = idx >> 16;
  const float* w = Wu + (size_t)n * UW_E;
  const float* yrow = Y + (size_t)n * NCOL + b * CIN;
  size_t base = (size_t)b * NN + n;
  float acc = bu[n * 32 + o];
  float x0 = src[(b * TT + t) * NN + n];
  acc += x0 * w[o] + yrow[0] * w[1056 + o];  // 1056 == k-stride of Wu
#pragma unroll 4
  for (int i = 1; i < CIN; ++i) {
    float zi = zr[base * 64 + i - 1];
    float hi = hin[base * HH + i - 1];
    acc += (zi * hi) * w[i * 32 + o] + yrow[i] * w[1056 + i * 32 + o];
  }
  float hc = tanhf(acc);
  float r = zr[base * 64 + 32 + o];
  float h = hin[base * HH + o];
  hout[base * HH + o] = r * h + (1.f - r) * hc;
}

// ---------------------------------------------------------------------------
// out[b,0,n,t] = sum_h h_last[b,n,h]*conv_W[t,h] + conv_b[t]
// ---------------------------------------------------------------------------
__global__ __launch_bounds__(256) void endconv_kernel(
    const float* __restrict__ h, const float* __restrict__ cw,
    const float* __restrict__ cb, float* __restrict__ out) {
  int idx = blockIdx.x * blockDim.x + threadIdx.x;
  if (idx >= BB * NN * TT) return;
  int t = idx % TT;
  int n = (idx / TT) & (NN - 1);
  int b = idx / (TT * NN);
  float acc = cb[t];
  size_t hb = ((size_t)b * NN + n) * HH;
#pragma unroll
  for (int j = 0; j < HH; ++j) acc += h[hb + j] * cw[t * HH + j];
  out[idx] = acc;
}

// ---------------------------------------------------------------------------
extern "C" void kernel_launch(void* const* d_in, const int* in_sizes, int n_in,
                              void* d_out, int out_size, void* d_ws,
                              size_t ws_size, hipStream_t stream) {
  const float* src = (const float*)d_in[0];
  const float* E   = (const float*)d_in[1];
  const float* gW  = (const float*)d_in[2];
  const float* gb  = (const float*)d_in[3];
  const float* uW  = (const float*)d_in[4];
  const float* ub  = (const float*)d_in[5];
  const float* cW  = (const float*)d_in[6];
  const float* cb  = (const float*)d_in[7];
  float* out = (float*)d_out;

  char* ws = (char*)d_ws;
  size_t off = 0;
  auto alloc = [&](size_t bytes) {
    void* p = ws + off;
    off = (off + bytes + 255) & ~(size_t)255;
    return p;
  };
  _Float16* A16 = (_Float16*)alloc((size_t)NN * NN * 2);
  _Float16* XT  = (_Float16*)alloc((size_t)NN * NCOL * 2);
  float* Ybuf = (float*)alloc((size_t)NN * NCOL * 4);
  float* Wg   = (float*)alloc((size_t)NN * GW_E * 4);
  float* Wu   = (float*)alloc((size_t)NN * UW_E * 4);
  float* bg   = (float*)alloc((size_t)NN * 64 * 4);
  float* bu   = (float*)alloc((size_t)NN * 32 * 4);
  float* h0   = (float*)alloc((size_t)BB * NN * HH * 4);
  float* h1   = (float*)alloc((size_t)BB * NN * HH * 4);
  float* zr   = (float*)alloc((size_t)BB * NN * 64 * 4);

  adj_kernel<<<NN, 256, 0, stream>>>(E, A16);
  params_kernel<<<NN, 256, 0, stream>>>(E, gW, gb, uW, ub, Wg, bg, Wu, bu);
  zero_kernel<<<(BB * NN * HH) / 256, 256, 0, stream>>>(h0, BB * NN * HH);

  const int xsBlocks = (NN * NCOL + 255) / 256;
  const dim3 gemmGrid(NN / 64, NCOL / 32);
  float* hin = h0;
  float* hout = h1;
  for (int t = 0; t < TT; ++t) {
    build_xs1<<<xsBlocks, 256, 0, stream>>>(src, hin, XT, t);
    gemm_kernel<<<gemmGrid, 128, 0, stream>>>(A16, XT, Ybuf);
    gate_kernel<<<(BB * NN * 64) / 256, 256, 0, stream>>>(src, hin, Ybuf, Wg,
                                                          bg, zr, t);
    build_xs2<<<xsBlocks, 256, 0, stream>>>(src, hin, zr, XT, t);
    gemm_kernel<<<gemmGrid, 128, 0, stream>>>(A16, XT, Ybuf);
    update_kernel<<<(BB * NN * HH) / 256, 256, 0, stream>>>(src, hin, zr, Ybuf,
                                                            Wu, bu, hout, t);
    float* tmp = hin; hin = hout; hout = tmp;
  }
  endconv_kernel<<<(BB * NN * TT) / 256, 256, 0, stream>>>(hin, cW, cb, out);
}
